// NeuroGNN_Encoder_30623116820797
// MI455X (gfx1250) — matrix operand order repaired and checked by reading, not compiled, not run on hardware
//
#include <hip/hip_runtime.h>
#include <hip/hip_bf16.h>

// ---------------- problem constants ----------------
#define B_    128
#define T_    64
#define N_    19
#define DIN_  100
#define H_    128
#define NH_   8
#define SEMIN_ 512
#define SEMD_ 168
#define CH_   128
#define OUT_  128
#define NODEF_ 296   // 128 + 168
#define H3_   384    // 3*H
#define KCAT_ 424    // NODEF_ + CH_
#define KCATP_ 448   // padded to 32
#define NODEFP_ 320  // 296 padded to 32

typedef __attribute__((ext_vector_type(16))) __bf16 v16bf;
typedef __attribute__((ext_vector_type(8)))  float  v8f;
typedef __attribute__((ext_vector_type(4)))  float  v4f;
typedef __attribute__((ext_vector_type(4)))  int    v4i;
typedef __attribute__((ext_vector_type(8)))  int    v8i;

#define WMMA_BF16(c, a, b) \
    (c) = __builtin_amdgcn_wmma_f32_16x16x32_bf16(false, (a), false, (b), (short)0, (c), false, false)

// ---------------- branch-free fragment loads from packed bf16 ----------------
// A 16x32: lane L row m=m0+(L&15); elements 0..7 -> k0+(L/16)*8 .. +7 (contig),
//          elements 8..15 -> k0+16+(L/16)*8 .. +7 (contig). Two 16B loads.
__device__ __forceinline__ v16bf frag_a_bf(const __bf16* __restrict__ A, int ldp,
                                           int m0, int k0, int lane) {
    int m = m0 + (lane & 15);
    int half = lane >> 4;
    const __bf16* p = A + (long)m * ldp + k0 + half * 8;
    v4i lo = *(const v4i*)p;
    v4i hi = *(const v4i*)(p + 16);
    v8i w = __builtin_shufflevector(lo, hi, 0, 1, 2, 3, 4, 5, 6, 7);
    return __builtin_bit_cast(v16bf, w);
}

// B 32x16 from row-major W[Nout,KP]: lane L col n=n0+(L&15); elements 0..7 ->
// k0+(L/16)*8 .. +7, elements 8..15 -> k0+16+(L/16)*8 .. +7. Two 16B loads.
__device__ __forceinline__ v16bf frag_b_bf(const __bf16* __restrict__ W, int ldp,
                                           int n0, int k0, int lane) {
    int n = n0 + (lane & 15);
    int half = lane >> 4;
    const __bf16* p = W + (long)n * ldp + k0 + half * 8;
    v4i lo = *(const v4i*)p;
    v4i hi = *(const v4i*)(p + 16);
    v8i w = __builtin_shufflevector(lo, hi, 0, 1, 2, 3, 4, 5, 6, 7);
    return __builtin_bit_cast(v16bf, w);
}

// A fragment from fp32 memory (contiguous vector loads, convert to bf16)
__device__ __forceinline__ v16bf frag_a_f32(const float* __restrict__ rowbase,
                                            int k0, int lane) {
    int half = lane >> 4;
    const float* p = rowbase + k0 + half * 8;
    v4f a0 = *(const v4f*)p;
    v4f a1 = *(const v4f*)(p + 4);
    v4f a2 = *(const v4f*)(p + 16);
    v4f a3 = *(const v4f*)(p + 20);
    v16bf f;
#pragma unroll
    for (int e = 0; e < 4; ++e) {
        f[e]      = (__bf16)a0[e];
        f[4 + e]  = (__bf16)a1[e];
        f[8 + e]  = (__bf16)a2[e];
        f[12 + e] = (__bf16)a3[e];
    }
    return f;
}

// ---------------- pack fp32 [R,C] -> bf16 [R,CP] zero padded ----------------
__global__ void k_pack_bf(const float* __restrict__ src, int R, int C,
                          __bf16* __restrict__ dst, int CP) {
    long idx = (long)blockIdx.x * blockDim.x + threadIdx.x;
    if (idx >= (long)R * CP) return;
    int c = (int)(idx % CP);
    long r = idx / CP;
    float v = (c < C) ? src[r * C + c] : 0.0f;
    dst[idx] = (__bf16)v;
}

// build Xcat bf16 [MB, 448] = [X(296) | Xg(128) | 0]
__global__ void k_pack_cat(const float* __restrict__ X, const float* __restrict__ Xg,
                           __bf16* __restrict__ dst) {
    long idx = (long)blockIdx.x * blockDim.x + threadIdx.x;
    const int MB = B_ * N_;
    if (idx >= (long)MB * KCATP_) return;
    int c = (int)(idx % KCATP_);
    long m = idx / KCATP_;
    float v;
    if (c < NODEF_)     v = X[m * NODEF_ + c];
    else if (c < KCAT_) v = Xg[m * CH_ + (c - NODEF_)];
    else                v = 0.0f;
    dst[idx] = (__bf16)v;
}

// ---------------- generic bf16 GEMM: act(A[M,KP] @ W[Nout,KP]^T + bias) ----------
// one wave computes a 16x64 strip (4 WMMA tiles sharing one A fragment)
__global__ void __launch_bounds__(256) k_gemm_bf(const __bf16* __restrict__ A, int ldp,
                                                 const __bf16* __restrict__ W, int wldp,
                                                 const float* __restrict__ bias,
                                                 float* __restrict__ outF, int ldcF,
                                                 __bf16* __restrict__ outB, int ldcB,
                                                 int M, int KP, int Nout, int relu) {
    int lane = threadIdx.x & 31;
    int wave = threadIdx.x >> 5;
    int groups = Nout >> 6;                       // strips of 64 columns
    long task = (long)blockIdx.x * 8 + wave;
    long tasks = (long)(M >> 4) * groups;
    if (task >= tasks) return;                    // wave-uniform
    int m0 = (int)(task / groups) << 4;
    int nb = (int)(task % groups) << 6;
    v8f c0 = {}, c1 = {}, c2 = {}, c3 = {};
    for (int k0 = 0; k0 < KP; k0 += 32) {
        v16bf a  = frag_a_bf(A, ldp, m0, k0, lane);
        v16bf b0 = frag_b_bf(W, wldp, nb,      k0, lane);
        v16bf b1 = frag_b_bf(W, wldp, nb + 16, k0, lane);
        v16bf b2 = frag_b_bf(W, wldp, nb + 32, k0, lane);
        v16bf b3 = frag_b_bf(W, wldp, nb + 48, k0, lane);
        WMMA_BF16(c0, a, b0);
        WMMA_BF16(c1, a, b1);
        WMMA_BF16(c2, a, b2);
        WMMA_BF16(c3, a, b3);
    }
    int nlo = lane & 15;
    int mb = m0 + ((lane >> 4) << 3);
    v8f acc[4] = {c0, c1, c2, c3};
#pragma unroll
    for (int q = 0; q < 4; ++q) {
        int n = nb + q * 16 + nlo;
        float bv = bias ? bias[n] : 0.0f;
#pragma unroll
        for (int v = 0; v < 8; ++v) {
            float x = acc[q][v] + bv;
            if (relu) x = fmaxf(x, 0.0f);
            if (outF) outF[(long)(mb + v) * ldcF + n] = x;
            if (outB) outB[(long)(mb + v) * ldcB + n] = (__bf16)x;
        }
    }
}

// ---------------- seq1 layer2, permuted bf16 store into xs[N,B,T,128] ----------------
__global__ void __launch_bounds__(256) k_seq1b(const __bf16* __restrict__ h1,
                                               const __bf16* __restrict__ W2,
                                               const float* __restrict__ b2,
                                               __bf16* __restrict__ xs) {
    int lane = threadIdx.x & 31;
    int wave = threadIdx.x >> 5;
    const int M = B_ * T_ * N_;
    long task = (long)blockIdx.x * 8 + wave;
    long tasks = (long)(M >> 4) * 2;              // Nout=128 -> 2 strips of 64
    if (task >= tasks) return;
    int m0 = (int)(task >> 1) << 4;
    int nb = (int)(task & 1) << 6;
    v8f c0 = {}, c1 = {}, c2 = {}, c3 = {};
    for (int k0 = 0; k0 < 64; k0 += 32) {
        v16bf a  = frag_a_bf(h1, 64, m0, k0, lane);
        v16bf b0 = frag_b_bf(W2, 64, nb,      k0, lane);
        v16bf b1 = frag_b_bf(W2, 64, nb + 16, k0, lane);
        v16bf b2 = frag_b_bf(W2, 64, nb + 32, k0, lane);
        v16bf b3 = frag_b_bf(W2, 64, nb + 48, k0, lane);
        WMMA_BF16(c0, a, b0);
        WMMA_BF16(c1, a, b1);
        WMMA_BF16(c2, a, b2);
        WMMA_BF16(c3, a, b3);
    }
    int nlo = lane & 15;
    int mb = m0 + ((lane >> 4) << 3);
    v8f acc[4] = {c0, c1, c2, c3};
#pragma unroll
    for (int q = 0; q < 4; ++q) {
        int col = nb + q * 16 + nlo;
        float bv = b2[col];
#pragma unroll
        for (int v = 0; v < 8; ++v) {
            int m = mb + v;
            int nn = m % N_;
            int bt = m / N_;
            int tt = bt % T_;
            int bb = bt / T_;
            float val = fmaxf(acc[q][v] + bv, 0.0f);
            xs[(((long)nn * B_ + bb) * T_ + tt) * H_ + col] = (__bf16)val;
        }
    }
}

// ---------------- GRU: h recurrence in registers, bf16 mirror in LDS ----------------
__global__ void __launch_bounds__(512) k_gru(const __bf16* __restrict__ xs,
                                             const __bf16* __restrict__ wih,
                                             const __bf16* __restrict__ whh,
                                             const float* __restrict__ bih,
                                             const float* __restrict__ bhh,
                                             float* __restrict__ outs,
                                             __bf16* __restrict__ outsbf,
                                             float* __restrict__ hT) {
    __shared__ __bf16 hbf[B_ * H_];                        // 32 KB
    int node = blockIdx.x;
    int lane = threadIdx.x & 31;
    int wave = threadIdx.x >> 5;                           // 0..15
    const __bf16* Wih = wih + (long)node * H3_ * H_;
    const __bf16* Whh = whh + (long)node * H3_ * H_;
    const float*  Bih = bih + (long)node * H3_;
    const float*  Bhh = bhh + (long)node * H3_;
    const __bf16* Xn  = xs  + (long)node * B_ * T_ * H_;

    float hreg[4][8];                                      // this thread's h elements
#pragma unroll
    for (int q = 0; q < 4; ++q)
#pragma unroll
        for (int v = 0; v < 8; ++v) hreg[q][v] = 0.0f;
    for (int i = threadIdx.x; i < B_ * H_; i += blockDim.x) hbf[i] = (__bf16)0.0f;
    __syncthreads();

    int nlo = lane & 15;
    int mhalf = (lane >> 4) << 3;

    for (int t = 0; t < T_; ++t) {
#pragma unroll
        for (int q = 0; q < 4; ++q) {
            int tile = wave * 4 + q;                       // 64 tiles = 8x8 grid
            int m0 = (tile >> 3) << 4;                     // batch rows
            int n0 = (tile & 7) << 4;                      // hidden cols
            v8f cir = {}, ciz = {}, cin = {}, chr_ = {}, chz = {}, chn = {};
            for (int k0 = 0; k0 < H_; k0 += 32) {
                v16bf ax = frag_a_bf(Xn + (long)t * H_, T_ * H_, m0, k0, lane);
                v16bf ah = frag_a_bf(hbf, H_, m0, k0, lane);        // ds reads
                v16bf br = frag_b_bf(Wih, H_, n0,          k0, lane);
                v16bf bz = frag_b_bf(Wih, H_, n0 + H_,     k0, lane);
                v16bf bn = frag_b_bf(Wih, H_, n0 + 2 * H_, k0, lane);
                WMMA_BF16(cir, ax, br);
                WMMA_BF16(ciz, ax, bz);
                WMMA_BF16(cin, ax, bn);
                v16bf hr = frag_b_bf(Whh, H_, n0,          k0, lane);
                v16bf hz = frag_b_bf(Whh, H_, n0 + H_,     k0, lane);
                v16bf hn = frag_b_bf(Whh, H_, n0 + 2 * H_, k0, lane);
                WMMA_BF16(chr_, ah, hr);
                WMMA_BF16(chz, ah, hz);
                WMMA_BF16(chn, ah, hn);
            }
            int j = n0 + nlo;
            float bir = Bih[j], biz = Bih[j + H_], bin = Bih[j + 2 * H_];
            float bhr = Bhh[j], bhz = Bhh[j + H_], bhn = Bhh[j + 2 * H_];
#pragma unroll
            for (int v = 0; v < 8; ++v) {
                float r  = 1.0f / (1.0f + __expf(-((cir[v] + bir) + (chr_[v] + bhr))));
                float z  = 1.0f / (1.0f + __expf(-((ciz[v] + biz) + (chz[v] + bhz))));
                float nn = tanhf((cin[v] + bin) + r * (chn[v] + bhn));
                hreg[q][v] = (1.0f - z) * nn + z * hreg[q][v];
            }
        }
        __syncthreads();
#pragma unroll
        for (int q = 0; q < 4; ++q) {
            int tile = wave * 4 + q;
            int m0 = (tile >> 3) << 4;
            int n0 = (tile & 7) << 4;
            int j = n0 + nlo;
#pragma unroll
            for (int v = 0; v < 8; ++v) {
                int b = m0 + mhalf + v;
                float hv = hreg[q][v];
                hbf[b * H_ + j] = (__bf16)hv;
                long oidx = (((long)node * B_ + b) * T_ + t) * H_ + j;
                outs[oidx] = hv;
                outsbf[oidx] = (__bf16)hv;
            }
        }
        __syncthreads();
    }
#pragma unroll
    for (int q = 0; q < 4; ++q) {
        int tile = wave * 4 + q;
        int m0 = (tile >> 3) << 4;
        int n0 = (tile & 7) << 4;
#pragma unroll
        for (int v = 0; v < 8; ++v)
            hT[((long)node * B_ + (m0 + mhalf + v)) * H_ + n0 + nlo] = hreg[q][v];
    }
}

// ---------------- Bahdanau time attention + residual + LayerNorm ----------------
__global__ void __launch_bounds__(256) k_time_attn(const float* __restrict__ outs,
                                                   const __bf16* __restrict__ outsbf,
                                                   const float* __restrict__ hT,
                                                   const __bf16* __restrict__ attn_wbf,
                                                   const float* __restrict__ attn_b,
                                                   const float* __restrict__ v_w,
                                                   const float* __restrict__ ln_g,
                                                   const float* __restrict__ ln_b,
                                                   __bf16* __restrict__ wrbf) {
    __shared__ float E[T_ * H_];      // 32 KB
    __shared__ float sc[T_];
    __shared__ float va[H_];
    __shared__ float stats[2];
    int node = blockIdx.x / B_;
    int b    = blockIdx.x % B_;
    const float*  ob  = outs   + ((long)node * B_ + b) * T_ * H_;
    const __bf16* obf = outsbf + ((long)node * B_ + b) * T_ * H_;
    const float*  hp  = hT     + ((long)node * B_ + b) * H_;
    int lane = threadIdx.x & 31;
    int wave = threadIdx.x >> 5;

    // E = tanh([hT | outs] @ attn_w^T + attn_b): M=64, K=256 (128 hT + 128 outs)
#pragma unroll
    for (int q = 0; q < 4; ++q) {
        int tile = wave * 4 + q;                 // 32 tiles = 4x8
        int m0 = (tile >> 3) << 4;
        int n0 = (tile & 7) << 4;
        v8f c = {};
        for (int k0 = 0; k0 < H_; k0 += 32) {    // hT half (rows identical)
            v16bf a = frag_a_f32(hp, k0, lane);
            v16bf bf = frag_b_bf(attn_wbf, 2 * H_, n0, k0, lane);
            WMMA_BF16(c, a, bf);
        }
        for (int k0 = 0; k0 < H_; k0 += 32) {    // outs half
            v16bf a = frag_a_bf(obf, H_, m0, k0, lane);
            v16bf bf = frag_b_bf(attn_wbf, 2 * H_, n0, H_ + k0, lane);
            WMMA_BF16(c, a, bf);
        }
        int col = n0 + (lane & 15);
        int mb  = m0 + ((lane >> 4) << 3);
        float bv = attn_b[col];
#pragma unroll
        for (int v = 0; v < 8; ++v) E[(mb + v) * H_ + col] = tanhf(c[v] + bv);
    }
    __syncthreads();
    if (threadIdx.x < T_) {
        int tt = threadIdx.x;
        float s = 0.0f;
        for (int j = 0; j < H_; ++j) s += E[tt * H_ + j] * v_w[j];
        sc[tt] = s;
    }
    __syncthreads();
    if (threadIdx.x == 0) {                      // deterministic softmax over T
        float mx = -1e30f;
        for (int tt = 0; tt < T_; ++tt) mx = fmaxf(mx, sc[tt]);
        float sum = 0.0f;
        for (int tt = 0; tt < T_; ++tt) { float e = __expf(sc[tt] - mx); sc[tt] = e; sum += e; }
        float inv = 1.0f / sum;
        for (int tt = 0; tt < T_; ++tt) sc[tt] *= inv;
    }
    __syncthreads();
    if (threadIdx.x < H_) {
        int h = threadIdx.x;
        float s = 0.0f;
        for (int tt = 0; tt < T_; ++tt) s += sc[tt] * ob[(long)tt * H_ + h];
        va[h] = s + hp[h];
    }
    __syncthreads();
    if (threadIdx.x == 0) {
        float mu = 0.0f;
        for (int h = 0; h < H_; ++h) mu += va[h];
        mu /= H_;
        float var = 0.0f;
        for (int h = 0; h < H_; ++h) { float d = va[h] - mu; var += d * d; }
        var /= H_;
        stats[0] = mu;
        stats[1] = rsqrtf(var + 1e-5f);
    }
    __syncthreads();
    if (threadIdx.x < H_) {
        int h = threadIdx.x;
        float v = (va[h] - stats[0]) * stats[1] * ln_g[h] + ln_b[h];
        wrbf[((long)b * N_ + node) * H_ + h] = (__bf16)v;
    }
}

// ---------------- MHA attention weights, per-batch head-averaged ----------------
__global__ void k_mha(const float* __restrict__ q, const float* __restrict__ k,
                      float* __restrict__ attb) {
    int b = blockIdx.x;
    int i = threadIdx.x;
    if (i >= N_) return;
    float accj[N_];
    for (int j = 0; j < N_; ++j) accj[j] = 0.0f;
    for (int h = 0; h < NH_; ++h) {
        float lg[N_];
        float mx = -1e30f;
        const float* qp = q + ((long)b * N_ + i) * H_ + h * 16;
        for (int j = 0; j < N_; ++j) {
            const float* kp = k + ((long)b * N_ + j) * H_ + h * 16;
            float s = 0.0f;
            for (int d = 0; d < 16; ++d) s += qp[d] * kp[d];
            s *= 0.25f;                                    // 1/sqrt(16)
            lg[j] = s;
            mx = fmaxf(mx, s);
        }
        float sum = 0.0f;
        for (int j = 0; j < N_; ++j) { lg[j] = __expf(lg[j] - mx); sum += lg[j]; }
        float inv = 1.0f / sum;
        for (int j = 0; j < N_; ++j) accj[j] += lg[j] * inv;
    }
    for (int j = 0; j < N_; ++j)
        attb[((long)b * N_ + i) * N_ + j] = accj[j] * (1.0f / NH_);
}

__global__ void k_att_reduce(const float* __restrict__ attb, float* __restrict__ att) {
    int idx = threadIdx.x;
    if (idx >= N_ * N_) return;
    float s = 0.0f;
    for (int b = 0; b < B_; ++b) s += attb[(long)b * N_ * N_ + idx];
    att[idx] = s * (1.0f / B_);
}

// ---------------- semantic embedding projection ----------------
__global__ void k_temb(const float* __restrict__ se, const float* __restrict__ w,
                       const float* __restrict__ bias, float* __restrict__ te) {
    int idx = blockIdx.x * blockDim.x + threadIdx.x;
    if (idx >= N_ * SEMD_) return;
    int n = idx / SEMD_, d = idx % SEMD_;
    float s = bias[d];
    const float* sp = se + (long)n * SEMIN_;
    const float* wp = w + (long)d * SEMIN_;
    for (int kk = 0; kk < SEMIN_; ++kk) s += sp[kk] * wp[kk];
    te[idx] = s;
}

// ---------------- adjacency build + mask + GCN normalization ----------------
__global__ void k_graph(const float* __restrict__ te, const float* __restrict__ dist,
                        const float* __restrict__ att, const float* __restrict__ alpha_p,
                        float* __restrict__ NrmOut) {
    __shared__ float nr[N_];
    __shared__ float Am[N_ * N_];
    __shared__ float rmax[N_];
    __shared__ float dinv[N_];
    int t = threadIdx.x;
    float alpha = *alpha_p;
    if (t < N_) {
        float s = 0.0f;
        const float* p = te + t * SEMD_;
        for (int d = 0; d < SEMD_; ++d) s += p[d] * p[d];
        nr[t] = sqrtf(s);
    }
    __syncthreads();
    if (t < N_ * N_) {
        int i = t / N_, j = t % N_;
        float dd = 0.0f;
        const float* pi = te + i * SEMD_;
        const float* pj = te + j * SEMD_;
        for (int d = 0; d < SEMD_; ++d) dd += pi[d] * pj[d];
        float ea = fabsf(dd / (nr[i] * nr[j]));
        Am[t] = (alpha * dist[t] + (1.0f - alpha) * ea) * att[t];
    }
    __syncthreads();
    if (t < N_) {
        float m = -1e30f;
        for (int j = 0; j < N_; ++j) m = fmaxf(m, Am[t * N_ + j]);
        rmax[t] = m;
    }
    __syncthreads();
    if (t < N_ * N_) {
        int i = t / N_;
        float a = Am[t];
        float msk = powf(a / rmax[i], 2.5f);
        Am[t] = (msk > 0.15f) ? a : 0.0f;
    }
    __syncthreads();
    if (t < N_) {                             // add remaining self loops
        int d = t * N_ + t;
        if (Am[d] == 0.0f) Am[d] = 1.0f;
    }
    __syncthreads();
    if (t < N_) {                             // deg over columns
        float s = 0.0f;
        for (int i = 0; i < N_; ++i) s += Am[i * N_ + t];
        dinv[t] = (s > 0.0f) ? rsqrtf(s) : 0.0f;
    }
    __syncthreads();
    if (t < N_ * N_) {
        int i = t / N_, j = t % N_;
        NrmOut[t] = dinv[i] * Am[t] * dinv[j];
    }
}

// ---------------- broadcast t_emb into X[:, :, 128:296] ----------------
__global__ void k_fill_temb(const float* __restrict__ te, float* __restrict__ X) {
    long idx = (long)blockIdx.x * blockDim.x + threadIdx.x;
    if (idx >= (long)B_ * N_ * SEMD_) return;
    int d = (int)(idx % SEMD_);
    long m = idx / SEMD_;
    int n = (int)(m % N_);
    X[m * NODEF_ + H_ + d] = te[n * SEMD_ + d];
}

// ---------------- GCN aggregation (writes fp32 + bf16) ----------------
__global__ void k_gcn_agg(const float* __restrict__ Y, const float* __restrict__ Nrm,
                          const float* __restrict__ bias, float* __restrict__ Z,
                          __bf16* __restrict__ Zbf) {
    long idx = (long)blockIdx.x * blockDim.x + threadIdx.x;
    if (idx >= (long)B_ * N_ * CH_) return;
    int f = (int)(idx % CH_);
    int j = (int)((idx / CH_) % N_);
    int b = (int)(idx / ((long)CH_ * N_));
    float s = bias[f];
    for (int i = 0; i < N_; ++i)
        s += Nrm[i * N_ + j] * Y[((long)b * N_ + i) * CH_ + f];
    s = fmaxf(s, 0.0f);
    Z[idx] = s;
    Zbf[idx] = (__bf16)s;
}

// =======================================================================
extern "C" void kernel_launch(void* const* d_in, const int* in_sizes, int n_in,
                              void* d_out, int out_size, void* d_ws, size_t ws_size,
                              hipStream_t stream) {
    (void)in_sizes; (void)n_in; (void)out_size; (void)ws_size;
    const float* x        = (const float*)d_in[0];
    const float* seq1_w1  = (const float*)d_in[1];
    const float* seq1_b1  = (const float*)d_in[2];
    const float* seq1_w2  = (const float*)d_in[3];
    const float* seq1_b2  = (const float*)d_in[4];
    const float* gru_wih  = (const float*)d_in[5];
    const float* gru_whh  = (const float*)d_in[6];
    const float* gru_bih  = (const float*)d_in[7];
    const float* gru_bhh  = (const float*)d_in[8];
    const float* attn_w   = (const float*)d_in[9];
    const float* attn_b   = (const float*)d_in[10];
    const float* v_w      = (const float*)d_in[11];
    const float* ln_g     = (const float*)d_in[12];
    const float* ln_b     = (const float*)d_in[13];
    const float* mha_in_w = (const float*)d_in[14];
    const float* mha_in_b = (const float*)d_in[15];
    const float* fc_ta_w  = (const float*)d_in[16];
    const float* fc_ta_b  = (const float*)d_in[17];
    const float* sem_embs = (const float*)d_in[18];
    const float* sem_w    = (const float*)d_in[19];
    const float* sem_b    = (const float*)d_in[20];
    const float* dist_adj = (const float*)d_in[21];
    const float* alpha_p  = (const float*)d_in[22];
    const float* gcn_w0   = (const float*)d_in[23];
    const float* gcn_b0   = (const float*)d_in[24];
    const float* gcn_w1   = (const float*)d_in[25];
    const float* gcn_b1   = (const float*)d_in[26];
    const float* gcn_w2   = (const float*)d_in[27];
    const float* gcn_b2   = (const float*)d_in[28];
    const float* fc_w     = (const float*)d_in[29];
    const float* fc_b     = (const float*)d_in[30];
    float* out = (float*)d_out;

    // ---- workspace (all allocations 256B aligned) ----
    float* ws = (float*)d_ws;
    size_t off = 0;
    auto allocf = [&](size_t n) {
        float* p = ws + off;
        off += (n + 63) & ~(size_t)63;
        return p;
    };
    auto allocb = [&](size_t n) {
        __bf16* p = (__bf16*)(ws + off);
        off += ((n + 1) / 2 + 63) & ~(size_t)63;
        return p;
    };
    const int M1 = B_ * T_ * N_;                          // 155648
    const int MB = B_ * N_;                               // 2432
    // fp32
    float* outsF = allocf((size_t)N_ * B_ * T_ * H_);
    float* hTb   = allocf((size_t)N_ * B_ * H_);
    float* qb    = allocf((size_t)MB * H_);
    float* kb    = allocf((size_t)MB * H_);
    float* attb  = allocf((size_t)B_ * N_ * N_);
    float* att   = allocf((size_t)N_ * N_);
    float* te    = allocf((size_t)N_ * SEMD_);
    float* nrm   = allocf((size_t)N_ * N_);
    float* X     = allocf((size_t)MB * NODEF_);
    float* Y     = allocf((size_t)MB * CH_);
    float* Xga   = allocf((size_t)MB * CH_);
    float* Xgb   = allocf((size_t)MB * CH_);
    // bf16 activations
    __bf16* xbf    = allocb((size_t)M1 * 128);            // x padded 100->128
    __bf16* h1bf   = allocb((size_t)M1 * 64);
    __bf16* xsbf   = allocb((size_t)N_ * B_ * T_ * H_);
    __bf16* outsB  = allocb((size_t)N_ * B_ * T_ * H_);
    __bf16* wrbf   = allocb((size_t)MB * H_);
    __bf16* Xbf    = allocb((size_t)MB * NODEFP_);
    __bf16* Xgabf  = allocb((size_t)MB * CH_);
    __bf16* Xgbbf  = allocb((size_t)MB * CH_);
    __bf16* Xcatbf = allocb((size_t)MB * KCATP_);
    // bf16 weights
    __bf16* w1bf   = allocb((size_t)64 * 128);
    __bf16* w2bf   = allocb((size_t)128 * 64);
    __bf16* wihbf  = allocb((size_t)N_ * H3_ * H_);
    __bf16* whhbf  = allocb((size_t)N_ * H3_ * H_);
    __bf16* awbf   = allocb((size_t)H_ * 2 * H_);
    __bf16* qkwbf  = allocb((size_t)256 * H_);
    __bf16* ftwbf  = allocb((size_t)128 * H_);
    __bf16* g0wbf  = allocb((size_t)CH_ * NODEFP_);
    __bf16* g1wbf  = allocb((size_t)CH_ * CH_);
    __bf16* g2wbf  = allocb((size_t)CH_ * CH_);
    __bf16* fcwbf  = allocb((size_t)OUT_ * KCATP_);

    auto packGrid = [](long n) { return (int)((n + 255) / 256); };

    // ---- pack weights + input ----
    k_pack_bf<<<packGrid((long)M1 * 128), 256, 0, stream>>>(x, M1, DIN_, xbf, 128);
    k_pack_bf<<<packGrid(64 * 128), 256, 0, stream>>>(seq1_w1, 64, DIN_, w1bf, 128);
    k_pack_bf<<<packGrid(128 * 64), 256, 0, stream>>>(seq1_w2, 128, 64, w2bf, 64);
    k_pack_bf<<<packGrid((long)N_ * H3_ * H_), 256, 0, stream>>>(gru_wih, N_ * H3_, H_, wihbf, H_);
    k_pack_bf<<<packGrid((long)N_ * H3_ * H_), 256, 0, stream>>>(gru_whh, N_ * H3_, H_, whhbf, H_);
    k_pack_bf<<<packGrid(H_ * 2 * H_), 256, 0, stream>>>(attn_w, H_, 2 * H_, awbf, 2 * H_);
    k_pack_bf<<<packGrid(256 * H_), 256, 0, stream>>>(mha_in_w, 256, H_, qkwbf, H_);
    k_pack_bf<<<packGrid(128 * H_), 256, 0, stream>>>(fc_ta_w, 128, H_, ftwbf, H_);
    k_pack_bf<<<packGrid(CH_ * NODEFP_), 256, 0, stream>>>(gcn_w0, CH_, NODEF_, g0wbf, NODEFP_);
    k_pack_bf<<<packGrid(CH_ * CH_), 256, 0, stream>>>(gcn_w1, CH_, CH_, g1wbf, CH_);
    k_pack_bf<<<packGrid(CH_ * CH_), 256, 0, stream>>>(gcn_w2, CH_, CH_, g2wbf, CH_);
    k_pack_bf<<<packGrid(OUT_ * KCATP_), 256, 0, stream>>>(fc_w, OUT_, KCAT_, fcwbf, KCATP_);

    // 1) seq1 layer1: h1bf = relu(x @ W1^T + b1)  (M1,128p)->(M1,64) bf16 out
    k_gemm_bf<<<(int)(((long)(M1 / 16) * 1 + 7) / 8), 256, 0, stream>>>(
        xbf, 128, w1bf, 128, seq1_b1, nullptr, 0, h1bf, 64, M1, 128, 64, 1);
    // 2) seq1 layer2 + permute -> xsbf[N,B,T,128]
    k_seq1b<<<(int)(((long)(M1 / 16) * 2 + 7) / 8), 256, 0, stream>>>(h1bf, w2bf, seq1_b2, xsbf);
    // 3) per-node GRU
    k_gru<<<N_, 512, 0, stream>>>(xsbf, wihbf, whhbf, gru_bih, gru_bhh, outsF, outsB, hTb);
    // 4) time attention + residual + LayerNorm -> wrbf
    k_time_attn<<<N_ * B_, 256, 0, stream>>>(outsF, outsB, hTb, awbf, attn_b, v_w,
                                             ln_g, ln_b, wrbf);
    // 5) Q/K projections (fp32 out for scalar MHA)
    k_gemm_bf<<<(MB / 16 * 2 + 7) / 8, 256, 0, stream>>>(
        wrbf, H_, qkwbf, H_, mha_in_b, qb, H_, nullptr, 0, MB, H_, H_, 0);
    k_gemm_bf<<<(MB / 16 * 2 + 7) / 8, 256, 0, stream>>>(
        wrbf, H_, qkwbf + 128 * H_, H_, mha_in_b + 128, kb, H_, nullptr, 0, MB, H_, H_, 0);
    // 6) head/batch-averaged attention
    k_mha<<<B_, 32, 0, stream>>>(qb, kb, attb);
    k_att_reduce<<<1, 384, 0, stream>>>(attb, att);
    // 7) semantic projection + graph build
    k_temb<<<packGrid(N_ * SEMD_), 256, 0, stream>>>(sem_embs, sem_w, sem_b, te);
    k_graph<<<1, 384, 0, stream>>>(te, dist_adj, att, alpha_p, nrm);
    // 8) X = [relu(wr @ fc_ta^T + b) | t_emb]
    k_gemm_bf<<<(MB / 16 * 2 + 7) / 8, 256, 0, stream>>>(
        wrbf, H_, ftwbf, H_, fc_ta_b, X, NODEF_, nullptr, 0, MB, H_, 128, 1);
    k_fill_temb<<<packGrid((long)MB * SEMD_), 256, 0, stream>>>(te, X);
    k_pack_bf<<<packGrid((long)MB * NODEFP_), 256, 0, stream>>>(X, MB, NODEF_, Xbf, NODEFP_);
    // 9) three GCN layers
    k_gemm_bf<<<(MB / 16 * 2 + 7) / 8, 256, 0, stream>>>(
        Xbf, NODEFP_, g0wbf, NODEFP_, nullptr, Y, CH_, nullptr, 0, MB, NODEFP_, CH_, 0);
    k_gcn_agg<<<packGrid((long)MB * CH_), 256, 0, stream>>>(Y, nrm, gcn_b0, Xga, Xgabf);
    k_gemm_bf<<<(MB / 16 * 2 + 7) / 8, 256, 0, stream>>>(
        Xgabf, CH_, g1wbf, CH_, nullptr, Y, CH_, nullptr, 0, MB, CH_, CH_, 0);
    k_gcn_agg<<<packGrid((long)MB * CH_), 256, 0, stream>>>(Y, nrm, gcn_b1, Xgb, Xgbbf);
    k_gemm_bf<<<(MB / 16 * 2 + 7) / 8, 256, 0, stream>>>(
        Xgbbf, CH_, g2wbf, CH_, nullptr, Y, CH_, nullptr, 0, MB, CH_, CH_, 0);
    k_gcn_agg<<<packGrid((long)MB * CH_), 256, 0, stream>>>(Y, nrm, gcn_b2, Xga, Xgabf);
    // 10) final FC on [X | Xg]
    k_pack_cat<<<packGrid((long)MB * KCATP_), 256, 0, stream>>>(X, Xga, Xcatbf);
    k_gemm_bf<<<(MB / 16 * 2 + 7) / 8, 256, 0, stream>>>(
        Xcatbf, KCATP_, fcwbf, KCATP_, fc_b, out, OUT_, nullptr, 0, MB, KCATP_, OUT_, 1);
}
